// GalacticNeuralNetwork_23373212025546
// MI455X (gfx1250) — matrix operand order repaired and checked by reading, not compiled, not run on hardware
//
#include <hip/hip_runtime.h>
#include <hip/hip_bf16.h>
#include <math.h>

// ---------------------------------------------------------------------------
// Problem constants (from the reference)
// ---------------------------------------------------------------------------
constexpr int Bb   = 4;
constexpr int Ss   = 1024;
constexpr int Dd   = 1024;
constexpr int NH   = 16;
constexpr int DH   = 64;
constexpr int FF   = 4096;
constexpr int NE   = 8;
constexpr int OUTD = 1024;
constexpr int NT   = Bb * Ss;          // 4096 tokens

typedef __bf16 bf16_t;
typedef __attribute__((ext_vector_type(16))) __bf16 v16bf;
typedef __attribute__((ext_vector_type(8)))  float  v8f;
typedef __attribute__((ext_vector_type(4)))  unsigned u32x4;   // native vec for NT loads

// GEMM tiling: 256 threads = 8 wave32; block tile 128x128x32.
// Waves arranged 4(M) x 2(N); each wave owns 32x64 = 2x4 WMMA 16x16 tiles.
constexpr int BM  = 128;
constexpr int BN  = 128;
constexpr int BK  = 32;
constexpr int BKP = 40;   // padded LDS row (16B-aligned, bank-spread)

#define CDIV(a, b) (((a) + (b) - 1) / (b))

static __device__ __forceinline__ float wave_sum(float v) {
  #pragma unroll
  for (int o = 16; o; o >>= 1) v += __shfl_xor(v, o, 32);
  return v;
}
static __device__ __forceinline__ float wave_max(float v) {
  #pragma unroll
  for (int o = 16; o; o >>= 1) v = fmaxf(v, __shfl_xor(v, o, 32));
  return v;
}

// ---------------------------------------------------------------------------
// fp32 -> bf16 conversion (grid-stride)
// ---------------------------------------------------------------------------
__global__ void k_cvt_bf16(const float* __restrict__ s, bf16_t* __restrict__ d, long n) {
  long i = (long)blockIdx.x * blockDim.x + threadIdx.x;
  long stride = (long)gridDim.x * blockDim.x;
  for (; i < n; i += stride) d[i] = (bf16_t)s[i];
}

// ---------------------------------------------------------------------------
// Sinusoidal relative position embedding, positions S-1 .. 0, bf16 out [S, D]
// ---------------------------------------------------------------------------
__global__ void k_pos_emb(bf16_t* __restrict__ pe) {
  int i = blockIdx.x * blockDim.x + threadIdx.x;
  if (i >= Ss * Dd) return;
  int srow = i / Dd, c = i % Dd;
  float pos = (float)(Ss - 1 - srow);
  int half = Dd / 2;
  int idx = (c < half) ? c : (c - half);
  // inv_freq = 10000^(-2*idx/D)
  float invf = __expf(-(float)(2 * idx) * (9.210340371976184f / (float)Dd));
  float a = pos * invf;
  pe[i] = (bf16_t)((c < half) ? __sinf(a) : __cosf(a));
}

// ---------------------------------------------------------------------------
// Split qkv (fp32 [NT,3D]) into bf16 q+u, q+v, k, v  (each [NT, D])
// ---------------------------------------------------------------------------
__global__ void k_qkv_prep(const float* __restrict__ qkv,
                           const float* __restrict__ ub, const float* __restrict__ vb,
                           bf16_t* __restrict__ qu, bf16_t* __restrict__ qv,
                           bf16_t* __restrict__ kk, bf16_t* __restrict__ vv) {
  long i = (long)blockIdx.x * blockDim.x + threadIdx.x;
  if (i >= (long)NT * Dd) return;
  long t = i / Dd; int d = (int)(i % Dd);
  const float* row = qkv + t * (3 * Dd);
  float q = row[d], k = row[Dd + d], v = row[2 * Dd + d];
  qu[i] = (bf16_t)(q + ub[d]);   // u_bias is [NH,DH] == flat [D]
  qv[i] = (bf16_t)(q + vb[d]);
  kk[i] = (bf16_t)k;
  vv[i] = (bf16_t)v;
}

// ---------------------------------------------------------------------------
// Fused flash-style softmax + P@V.
//   scores : fp32 [B*NH, S, S], pre-scaled by 1/sqrt(DH); entries above the
//            diagonal are garbage and masked here.
//   v      : bf16 [B, S, NH*DH]
//   ctx    : bf16 [B, S, NH*DH]   (normalized output, written directly)
// Grid: (S/128, B*NH); 256 threads = 8 wave32, wave w owns query rows
// [qt*128 + 16w , +16). Online (max,sum) softmax over 32-wide key chunks,
// P packed to bf16 A-fragments in registers, P@V via WMMA.
// ---------------------------------------------------------------------------
__global__ void __launch_bounds__(256) k_flash_pv(
    const float* __restrict__ scores, const bf16_t* __restrict__ v,
    bf16_t* __restrict__ ctx) {
  const int qt = blockIdx.x;
  const int z  = blockIdx.y;
  const int b  = z / NH, hh = z % NH;
  const float* srow = scores + (long)z * Ss * Ss;
  const bf16_t* vbase = v + (long)b * Ss * Dd + hh * DH;
  bf16_t* obase = ctx + (long)b * Ss * Dd + hh * DH;

  const int tid = threadIdx.x, lane = tid & 31, wid = tid >> 5;
  const int l15 = lane & 15, half = lane >> 4;
  const int myrow = qt * 128 + wid * 16 + l15;   // query row for A-layout lane

  __shared__ float  Stile[128][36];   // 32 score cols per chunk (+pad)
  __shared__ bf16_t Vs[64][BKP];      // V chunk, n-major [dh][k]

  union Frag { v16bf v; unsigned u[8]; };
  union H2 { __bf16 h[2]; unsigned u; };

  v8f acc[4];
  #pragma unroll
  for (int tn = 0; tn < 4; tn++)
    #pragma unroll
    for (int r = 0; r < 8; r++) acc[tn][r] = 0.f;
  float m = -3.0e38f, l = 0.f;

  const int nchunks = (qt + 1) * 4;   // keys 0 .. (qt+1)*128-1
  for (int c = 0; c < nchunks; c++) {
    const int j0 = c * 32;
    // ---- stage score chunk: 128 rows x 32 cols fp32 (stream-once -> NT) ----
    #pragma unroll
    for (int i = 0; i < 4; i++) {
      int chunk = tid + i * 256;                 // 1024 chunks of 4 floats
      int row = chunk >> 3, c4 = (chunk & 7) << 2;
      u32x4 d = __builtin_nontemporal_load(
          (const u32x4*)(srow + (long)(qt * 128 + row) * Ss + j0 + c4));
      *(u32x4*)(&Stile[row][c4]) = d;
    }
    // ---- stage V chunk: 32 (k) x 64 (dh), transposed to [dh][k] ----
    {
      int kr = tid >> 3, c8 = (tid & 7) << 3;
      bf16_t tmp[8];
      *(uint4*)tmp = *(const uint4*)(vbase + (long)(j0 + kr) * Dd + c8);
      #pragma unroll
      for (int j = 0; j < 8; j++) Vs[c8 + j][kr] = tmp[j];
    }
    __syncthreads();

    // ---- per-lane scores in A-fragment order ----
    float sv[16];
    #pragma unroll
    for (int i2 = 0; i2 < 8; i2++) {
      int kb = ((i2 >> 2) << 4) + (half << 3) + ((i2 & 3) << 1);
      sv[2 * i2]     = Stile[wid * 16 + l15][kb];
      sv[2 * i2 + 1] = Stile[wid * 16 + l15][kb + 1];
    }
    // ---- online softmax update ----
    float cm = -3.0e38f;
    #pragma unroll
    for (int e = 0; e < 16; e++) {
      int i2 = e >> 1;
      int j = j0 + ((i2 >> 2) << 4) + (half << 3) + ((i2 & 3) << 1) + (e & 1);
      sv[e] = (j <= myrow) ? sv[e] : -3.0e38f;
      cm = fmaxf(cm, sv[e]);
    }
    cm = fmaxf(cm, __shfl_xor(cm, 16, 32));
    float mnew = fmaxf(m, cm);
    float scale = __expf(m - mnew);
    m = mnew;
    l *= scale;
    float cs = 0.f, pv[16];
    #pragma unroll
    for (int e = 0; e < 16; e++) {
      float p = (sv[e] > -1.0e37f) ? __expf(sv[e] - mnew) : 0.f;
      pv[e] = p; cs += p;
    }
    cs += __shfl_xor(cs, 16, 32);
    l += cs;
    // rescale accumulators by per-row factor (row r+8*half lives in lane r+8*half)
    float fr[8];
    #pragma unroll
    for (int r = 0; r < 8; r++) fr[r] = __shfl(scale, r + 8 * half, 32);
    #pragma unroll
    for (int tn = 0; tn < 4; tn++)
      #pragma unroll
      for (int r = 0; r < 8; r++) acc[tn][r] *= fr[r];
    // ---- pack P -> bf16 A fragment ----
    Frag af;
    #pragma unroll
    for (int i2 = 0; i2 < 8; i2++) {
      H2 t;
      t.h[0] = (bf16_t)pv[2 * i2];
      t.h[1] = (bf16_t)pv[2 * i2 + 1];
      af.u[i2] = t.u;
    }
    // ---- B fragments from Vs and 4 WMMAs ----
    #pragma unroll
    for (int tn = 0; tn < 4; tn++) {
      Frag bfx;
      int n = tn * 16 + l15;
      #pragma unroll
      for (int i = 0; i < 8; i++)
        bfx.u[i] = *(const unsigned*)(&Vs[n][((i >> 2) << 4) + (half << 3) + ((i & 3) << 1)]);
      acc[tn] = __builtin_amdgcn_wmma_f32_16x16x32_bf16(
          false, af.v, false, bfx.v, (short)0, acc[tn], false, false);
    }
    __syncthreads();
  }

  // ---- final normalize and store bf16 ctx ----
  float inv = 1.f / l;
  float fr[8];
  #pragma unroll
  for (int r = 0; r < 8; r++) fr[r] = __shfl(inv, r + 8 * half, 32);
  #pragma unroll
  for (int tn = 0; tn < 4; tn++) {
    int dh = tn * 16 + l15;
    #pragma unroll
    for (int r = 0; r < 8; r++) {
      int qrow = qt * 128 + wid * 16 + r + 8 * half;
      obase[(long)qrow * Dd + dh] = (bf16_t)(acc[tn][r] * fr[r]);
    }
  }
}

// ---------------------------------------------------------------------------
// out = LayerNorm(x + y); writes fp32 (optional) and bf16 (optional)
// ---------------------------------------------------------------------------
__global__ void __launch_bounds__(256) k_ln_res(
    const float* __restrict__ x, const float* __restrict__ y,
    const float* __restrict__ g, const float* __restrict__ b,
    float* __restrict__ of, bf16_t* __restrict__ ob) {
  long row = blockIdx.x;
  int tid = threadIdx.x;
  const float* xr = x + row * Dd;
  const float* yr = y + row * Dd;
  __shared__ float r1[8], r2[8];

  float v[4];
  float s = 0.f, ss = 0.f;
  #pragma unroll
  for (int i = 0; i < 4; i++) {
    int d = tid + i * 256;
    float t = xr[d] + yr[d];
    v[i] = t; s += t; ss += t * t;
  }
  s = wave_sum(s); ss = wave_sum(ss);
  if ((tid & 31) == 0) { r1[tid >> 5] = s; r2[tid >> 5] = ss; }
  __syncthreads();
  s  = r1[0] + r1[1] + r1[2] + r1[3] + r1[4] + r1[5] + r1[6] + r1[7];
  ss = r2[0] + r2[1] + r2[2] + r2[3] + r2[4] + r2[5] + r2[6] + r2[7];
  float mean = s * (1.f / Dd);
  float var  = ss * (1.f / Dd) - mean * mean;
  float rs   = rsqrtf(var + 1e-5f);
  #pragma unroll
  for (int i = 0; i < 4; i++) {
    int d = tid + i * 256;
    float o = (v[i] - mean) * rs * g[d] + b[d];
    if (of) of[row * Dd + d] = o;
    if (ob) ob[row * Dd + d] = (bf16_t)o;
  }
}

// ---------------------------------------------------------------------------
// Top-1 router: probs = softmax(h @ Wg); gate = max prob, expert = argmax.
// ---------------------------------------------------------------------------
__global__ void __launch_bounds__(256) k_router(
    const float* __restrict__ h, const float* __restrict__ Wg,
    float* __restrict__ gate, int* __restrict__ expert, int* __restrict__ counts) {
  int t = blockIdx.x;
  int tid = threadIdx.x;
  float acc[NE];
  #pragma unroll
  for (int e = 0; e < NE; e++) acc[e] = 0.f;
  for (int i = 0; i < 4; i++) {
    int d = tid + i * 256;
    float xv = h[(long)t * Dd + d];
    #pragma unroll
    for (int e = 0; e < NE; e++) acc[e] += xv * Wg[d * NE + e];
  }
  __shared__ float red[8][NE];
  #pragma unroll
  for (int e = 0; e < NE; e++) acc[e] = wave_sum(acc[e]);
  if ((tid & 31) == 0) {
    #pragma unroll
    for (int e = 0; e < NE; e++) red[tid >> 5][e] = acc[e];
  }
  __syncthreads();
  if (tid == 0) {
    float lg[NE];
    #pragma unroll
    for (int e = 0; e < NE; e++)
      lg[e] = red[0][e] + red[1][e] + red[2][e] + red[3][e] +
              red[4][e] + red[5][e] + red[6][e] + red[7][e];
    float mx = lg[0]; int am = 0;
    #pragma unroll
    for (int e = 1; e < NE; e++) if (lg[e] > mx) { mx = lg[e]; am = e; }
    float s = 0.f;
    #pragma unroll
    for (int e = 0; e < NE; e++) s += __expf(lg[e] - mx);
    gate[t] = 1.f / s;           // softmax value at the argmax
    expert[t] = am;
    atomicAdd(&counts[am], 1);
  }
}

__global__ void k_zero_i32(int* p, int n) {
  int i = blockIdx.x * blockDim.x + threadIdx.x;
  if (i < n) p[i] = 0;
}

__global__ void k_scan(const int* __restrict__ counts, int* __restrict__ offs) {
  if (blockIdx.x == 0 && threadIdx.x == 0) {
    int a = 0;
    for (int e = 0; e < NE; e++) { offs[e] = a; a += counts[e]; }
    offs[NE] = a;
  }
}

__global__ void k_scatter(const int* __restrict__ expert, const int* __restrict__ offs,
                          int* __restrict__ cursor, int* __restrict__ perm) {
  int t = blockIdx.x * blockDim.x + threadIdx.x;
  if (t >= NT) return;
  int e = expert[t];
  int pos = atomicAdd(&cursor[e], 1);
  perm[offs[e] + pos] = t;
}

// ---------------------------------------------------------------------------
// WMMA GEMM:  C = epilogue(alpha * A @ B + bias)
//   BT=1      : B given as Bt[N][K] row-major (compute A @ Bt^T)
//   GATHER_A=1: A rows gathered through perm[] (expert dispatch)
//   BCHK=1    : enable M/N bounds handling (data-dependent expert batches)
//   EP: 0 = f32 store          1 = bf16 store        2 = relu -> bf16 store
//       3 = rel-shift += into f32 scores (k = n-(N-1)+m)
//       5 = scatter by perm with gate scaling (MoE combine)
// ---------------------------------------------------------------------------
template <int BT, int GATHER_A, int EP, int BCHK>
__global__ void __launch_bounds__(256) k_gemm(
    const bf16_t* __restrict__ A0, long lda, long as1, long as2,
    const bf16_t* __restrict__ B0, long ldb, long bs1, long bs2,
    void* __restrict__ C0, long ldc, long cs1, long cs2,
    const float* __restrict__ bias0, long bias_stride,
    int zdiv, int M, int N, int K, float alpha,
    const int* __restrict__ perm, const float* __restrict__ gatev,
    const int* __restrict__ eoffs) {
  const int z = blockIdx.z;
  const int z1 = z / zdiv, z2 = z % zdiv;
  const bf16_t* A  = A0 + (long)z1 * as1 + (long)z2 * as2;
  const bf16_t* Bm = B0 + (long)z1 * bs1 + (long)z2 * bs2;
  const long coff  = (long)z1 * cs1 + (long)z2 * cs2;
  const float* bias = bias0 ? (bias0 + (long)z * bias_stride) : nullptr;

  int Mvalid = M, rowbase = 0;
  const int* permz = nullptr;
  if (eoffs) {
    int eo = eoffs[z];
    rowbase = eo;
    Mvalid = eoffs[z + 1] - eo;
    if (perm) permz = perm + eo;
  }
  const int m0b = blockIdx.y * BM;
  const int n0b = blockIdx.x * BN;
  if (BCHK && m0b >= Mvalid) return;

  __shared__ bf16_t As[BM][BKP];   // A tile, row-major [m][k]
  __shared__ bf16_t Bs[BN][BKP];   // B tile, n-major  [n][k]

  const int tid = threadIdx.x;
  const int lane = tid & 31;
  const int wid = tid >> 5;
  const int wm0 = (wid >> 1) * 32;
  const int wn0 = (wid & 1) * 64;
  const int l15 = lane & 15;
  const int half = lane >> 4;

  v8f acc[2][4];
  #pragma unroll
  for (int i = 0; i < 2; i++)
    #pragma unroll
    for (int j = 0; j < 4; j++)
      #pragma unroll
      for (int r = 0; r < 8; r++) acc[i][j][r] = 0.f;

  union Frag { v16bf v; unsigned u[8]; };

  for (int k0 = 0; k0 < K; k0 += BK) {
    // ---- stage A tile (512 chunks of 8 bf16, 2 per thread) ----
    #pragma unroll
    for (int i = 0; i < 2; i++) {
      int chunk = tid + i * 256;
      int row = chunk >> 2, c16 = (chunk & 3) << 3;
      int r = m0b + row;
      int rc = BCHK ? ((r < Mvalid) ? r : (Mvalid - 1)) : r;
      const bf16_t* arow;
      if (GATHER_A) arow = A + (long)permz[rc] * lda;
      else          arow = A + (long)(rowbase + rc) * lda;
      *(uint4*)(&As[row][c16]) = *(const uint4*)(arow + k0 + c16);
    }
    // ---- stage B tile ----
    if (BT) {
      #pragma unroll
      for (int i = 0; i < 2; i++) {
        int chunk = tid + i * 256;
        int row = chunk >> 2, c16 = (chunk & 3) << 3;
        int n = n0b + row;
        int nc = BCHK ? ((n < N) ? n : (N - 1)) : n;
        *(uint4*)(&Bs[row][c16]) = *(const uint4*)(Bm + (long)nc * ldb + k0 + c16);
      }
    } else {
      #pragma unroll
      for (int i = 0; i < 2; i++) {
        int chunk = tid + i * 256;
        int kr = chunk >> 4, c8 = (chunk & 15) << 3;
        int n = n0b + c8;
        bf16_t tmp[8];
        if (!BCHK || n < N) {
          *(uint4*)tmp = *(const uint4*)(Bm + (long)(k0 + kr) * ldb + n);
        } else {
          #pragma unroll
          for (int j = 0; j < 8; j++) tmp[j] = (bf16_t)0.f;
        }
        #pragma unroll
        for (int j = 0; j < 8; j++) Bs[c8 + j][kr] = tmp[j];  // transpose to [n][k]
      }
    }
    __syncthreads();

    // Prefetch next K-tile of A into cache (gfx1250 global_prefetch)
    if (!GATHER_A && (k0 + BK < K)) {
      const bf16_t* pa =
          A + (long)(rowbase + m0b + (tid >> 1)) * lda + (k0 + BK) + ((tid & 1) << 4);
      __builtin_prefetch(pa, 0, 1);
    }

    // ---- load fragments (per documented wave32 16-bit WMMA layouts) ----
    Frag af[2], bfx[4];
    #pragma unroll
    for (int tm = 0; tm < 2; tm++) {
      int mm = wm0 + tm * 16 + l15;
      #pragma unroll
      for (int i = 0; i < 8; i++) {
        int kb = ((i >> 2) << 4) + (half << 3) + ((i & 3) << 1);
        af[tm].u[i] = *(const unsigned*)(&As[mm][kb]);
      }
    }
    #pragma unroll
    for (int tn = 0; tn < 4; tn++) {
      int n = wn0 + tn * 16 + l15;
      #pragma unroll
      for (int i = 0; i < 8; i++) {
        int kb = ((i >> 2) << 4) + (half << 3) + ((i & 3) << 1);
        bfx[tn].u[i] = *(const unsigned*)(&Bs[n][kb]);
      }
    }
    // ---- 8 WMMA per wave per K-step ----
    #pragma unroll
    for (int tm = 0; tm < 2; tm++)
      #pragma unroll
      for (int tn = 0; tn < 4; tn++)
        acc[tm][tn] = __builtin_amdgcn_wmma_f32_16x16x32_bf16(
            false, af[tm].v, false, bfx[tn].v, (short)0, acc[tm][tn], false, false);
    __syncthreads();
  }

  // ---- epilogue ----
  float*  Cf = (float*)C0 + coff;
  bf16_t* Cb = (bf16_t*)C0 + coff;
  #pragma unroll
  for (int tm = 0; tm < 2; tm++) {
    #pragma unroll
    for (int tn = 0; tn < 4; tn++) {
      int n = n0b + wn0 + tn * 16 + l15;
      bool nok = BCHK ? (n < N) : true;
      float bv = (bias && nok) ? bias[n] : 0.f;
      #pragma unroll
      for (int r = 0; r < 8; r++) {
        int mi = m0b + wm0 + tm * 16 + (half << 3) + r;
        bool mok = BCHK ? (mi < Mvalid) : true;
        float v = acc[tm][tn][r] * alpha + bv;
        if (EP == 0) {
          if (mok && nok) Cf[(long)(rowbase + mi) * ldc + n] = v;
        } else if (EP == 1) {
          if (mok && nok) Cb[(long)(rowbase + mi) * ldc + n] = (bf16_t)v;
        } else if (EP == 2) {
          if (mok && nok) Cb[(long)(rowbase + mi) * ldc + n] = (bf16_t)fmaxf(v, 0.f);
        } else if (EP == 3) {
          int kc = n - (N - 1) + mi;          // rel_shift: j -> k
          if (mok && nok && kc >= 0) Cf[(long)mi * ldc + kc] += v;
        } else if (EP == 5) {
          if (mok && nok) {
            int t = permz[mi];
            Cf[(long)t * ldc + n] = gatev[t] * v;
          }
        }
      }
    }
  }
}

// ---------------------------------------------------------------------------
// Host-side orchestration
// ---------------------------------------------------------------------------
extern "C" void kernel_launch(void* const* d_in, const int* in_sizes, int n_in,
                              void* d_out, int out_size, void* d_ws, size_t ws_size,
                              hipStream_t stream) {
  (void)in_sizes; (void)n_in; (void)out_size; (void)ws_size;
  const float* x     = (const float*)d_in[0];
  const float* Wqkv  = (const float*)d_in[1];
  const float* Wo    = (const float*)d_in[2];
  const float* Wr    = (const float*)d_in[3];
  const float* ub    = (const float*)d_in[4];
  const float* vb    = (const float*)d_in[5];
  const float* ln1g  = (const float*)d_in[6];
  const float* ln1b  = (const float*)d_in[7];
  const float* Wff1  = (const float*)d_in[8];
  const float* bff1  = (const float*)d_in[9];
  const float* Wff2  = (const float*)d_in[10];
  const float* bff2  = (const float*)d_in[11];
  const float* ln2g  = (const float*)d_in[12];
  const float* ln2b  = (const float*)d_in[13];
  const float* Wg    = (const float*)d_in[14];
  const float* We1   = (const float*)d_in[15];
  const float* be1   = (const float*)d_in[16];
  const float* We2   = (const float*)d_in[17];
  const float* be2   = (const float*)d_in[18];
  const float* ln3g  = (const float*)d_in[19];
  const float* ln3b  = (const float*)d_in[20];
  const float* Wout  = (const float*)d_in[21];
  float* out = (float*)d_out;

  // bump allocator over d_ws (~720 MB)
  char* w = (char*)d_ws;
  auto take = [&](size_t bytes) -> char* {
    char* p = w;
    w += (bytes + 255) & ~(size_t)255;
    return p;
  };
  bf16_t* wqkvb = (bf16_t*)take((size_t)Dd * 3 * Dd * 2);
  bf16_t* wob   = (bf16_t*)take((size_t)Dd * Dd * 2);
  bf16_t* wrb   = (bf16_t*)take((size_t)Dd * Dd * 2);
  bf16_t* wff1b = (bf16_t*)take((size_t)Dd * FF * 2);
  bf16_t* wff2b = (bf16_t*)take((size_t)FF * Dd * 2);
  bf16_t* woutb = (bf16_t*)take((size_t)Dd * OUTD * 2);
  bf16_t* we1b  = (bf16_t*)take((size_t)NE * Dd * FF * 2);
  bf16_t* we2b  = (bf16_t*)take((size_t)NE * FF * Dd * 2);
  bf16_t* xb    = (bf16_t*)take((size_t)NT * Dd * 2);
  bf16_t* peb   = (bf16_t*)take((size_t)Ss * Dd * 2);
  float*  qkvf  = (float*)take((size_t)NT * 3 * Dd * 4);
  bf16_t* qub   = (bf16_t*)take((size_t)NT * Dd * 2);
  bf16_t* qvb   = (bf16_t*)take((size_t)NT * Dd * 2);
  bf16_t* kb    = (bf16_t*)take((size_t)NT * Dd * 2);
  bf16_t* vvb   = (bf16_t*)take((size_t)NT * Dd * 2);
  bf16_t* rb    = (bf16_t*)take((size_t)Ss * Dd * 2);
  float*  scores= (float*)take((size_t)Bb * NH * Ss * Ss * 4);
  bf16_t* ctxb  = (bf16_t*)take((size_t)NT * Dd * 2);
  float*  aof   = (float*)take((size_t)NT * Dd * 4);
  float*  h1f   = (float*)take((size_t)NT * Dd * 4);
  bf16_t* h1b   = (bf16_t*)take((size_t)NT * Dd * 2);
  bf16_t* ff1b  = (bf16_t*)take((size_t)NT * FF * 2);
  float*  ff2f  = (float*)take((size_t)NT * Dd * 4);
  float*  h2f   = (float*)take((size_t)NT * Dd * 4);
  bf16_t* h2b   = (bf16_t*)take((size_t)NT * Dd * 2);
  float*  gate  = (float*)take((size_t)NT * 4);
  int*    expid = (int*)take((size_t)NT * 4);
  int*    counts= (int*)take(64);            // counts[8] + cursor[8]
  int*    cursor= counts + 8;
  int*    offs  = (int*)take(64);            // offs[9]
  int*    perm  = (int*)take((size_t)NT * 4);
  bf16_t* hidb  = (bf16_t*)take((size_t)NT * FF * 2);
  float*  moef  = (float*)take((size_t)NT * Dd * 4);
  bf16_t* h3b   = (bf16_t*)take((size_t)NT * Dd * 2);

  auto cvt = [&](const float* s, bf16_t* d, long n) {
    k_cvt_bf16<<<dim3(2048), dim3(256), 0, stream>>>(s, d, n);
  };

  // ---- precision conversion of activations and weights ----
  cvt(x, xb, (long)NT * Dd);
  cvt(Wqkv, wqkvb, (long)Dd * 3 * Dd);
  cvt(Wo, wob, (long)Dd * Dd);
  cvt(Wr, wrb, (long)Dd * Dd);
  cvt(Wff1, wff1b, (long)Dd * FF);
  cvt(Wff2, wff2b, (long)FF * Dd);
  cvt(We1, we1b, (long)NE * Dd * FF);
  cvt(We2, we2b, (long)NE * FF * Dd);
  cvt(Wout, woutb, (long)Dd * OUTD);
  k_pos_emb<<<dim3(CDIV(Ss * Dd, 256)), dim3(256), 0, stream>>>(peb);

  const long SD = (long)Ss * Dd;        // per-batch stride of [S, NH*DH]
  const long SS = (long)Ss * Ss;        // per-head score stride

  // ---- qkv = x @ Wqkv ----
  k_gemm<0, 0, 0, 0><<<dim3(CDIV(3 * Dd, BN), CDIV(NT, BM), 1), 256, 0, stream>>>(
      xb, Dd, 0, 0, wqkvb, 3 * Dd, 0, 0, qkvf, 3 * Dd, 0, 0,
      nullptr, 0, 1, NT, 3 * Dd, Dd, 1.f, nullptr, nullptr, nullptr);
  k_qkv_prep<<<dim3(CDIV(NT * Dd, 256)), dim3(256), 0, stream>>>(
      qkvf, ub, vb, qub, qvb, kb, vvb);

  // ---- r = pe @ Wr (bf16 out) ----
  k_gemm<0, 0, 1, 0><<<dim3(CDIV(Dd, BN), CDIV(Ss, BM), 1), 256, 0, stream>>>(
      peb, Dd, 0, 0, wrb, Dd, 0, 0, rb, Dd, 0, 0,
      nullptr, 0, 1, Ss, Dd, Dd, 1.f, nullptr, nullptr, nullptr);

  // ---- scores = 0.125 * (q+u) @ k^T, batched over (b,h) ----
  k_gemm<1, 0, 0, 0><<<dim3(CDIV(Ss, BN), CDIV(Ss, BM), Bb * NH), 256, 0, stream>>>(
      qub, Dd, SD, DH, kb, Dd, SD, DH, scores, Ss, (long)NH * SS, SS,
      nullptr, 0, NH, Ss, Ss, DH, 0.125f, nullptr, nullptr, nullptr);

  // ---- scores += rel_shift(0.125 * (q+v) @ r^T) ----
  k_gemm<1, 0, 3, 0><<<dim3(CDIV(Ss, BN), CDIV(Ss, BM), Bb * NH), 256, 0, stream>>>(
      qvb, Dd, SD, DH, rb, Dd, 0, DH, scores, Ss, (long)NH * SS, SS,
      nullptr, 0, NH, Ss, Ss, DH, 0.125f, nullptr, nullptr, nullptr);

  // ---- fused causal softmax + P@V -> bf16 ctx ----
  k_flash_pv<<<dim3(Ss / 128, Bb * NH), 256, 0, stream>>>(scores, vvb, ctxb);

  // ---- attn out = ctx @ Wo ; h1 = LN(x + attn_out) ----
  k_gemm<0, 0, 0, 0><<<dim3(CDIV(Dd, BN), CDIV(NT, BM), 1), 256, 0, stream>>>(
      ctxb, Dd, 0, 0, wob, Dd, 0, 0, aof, Dd, 0, 0,
      nullptr, 0, 1, NT, Dd, Dd, 1.f, nullptr, nullptr, nullptr);
  k_ln_res<<<dim3(NT), dim3(256), 0, stream>>>(x, aof, ln1g, ln1b, h1f, h1b);

  // ---- FFN: ff1 = relu(h1 @ Wff1 + b1) ; ff2 = ff1 @ Wff2 + b2 ----
  k_gemm<0, 0, 2, 0><<<dim3(CDIV(FF, BN), CDIV(NT, BM), 1), 256, 0, stream>>>(
      h1b, Dd, 0, 0, wff1b, FF, 0, 0, ff1b, FF, 0, 0,
      bff1, 0, 1, NT, FF, Dd, 1.f, nullptr, nullptr, nullptr);
  k_gemm<0, 0, 0, 0><<<dim3(CDIV(Dd, BN), CDIV(NT, BM), 1), 256, 0, stream>>>(
      ff1b, FF, 0, 0, wff2b, Dd, 0, 0, ff2f, Dd, 0, 0,
      bff2, 0, 1, NT, Dd, FF, 1.f, nullptr, nullptr, nullptr);
  k_ln_res<<<dim3(NT), dim3(256), 0, stream>>>(h1f, ff2f, ln2g, ln2b, h2f, h2b);

  // ---- MoE: top-1 route, group tokens by expert, expert GEMMs, combine ----
  k_zero_i32<<<dim3(1), dim3(32), 0, stream>>>(counts, 16);
  k_router<<<dim3(NT), dim3(256), 0, stream>>>(h2f, Wg, gate, expid, counts);
  k_scan<<<dim3(1), dim3(32), 0, stream>>>(counts, offs);
  k_scatter<<<dim3(CDIV(NT, 256)), dim3(256), 0, stream>>>(expid, offs, cursor, perm);

  // hid[slot] = relu(h2[perm[slot]] @ We1[e] + be1[e])   (gather-A, grouped by expert)
  k_gemm<0, 1, 2, 1><<<dim3(CDIV(FF, BN), CDIV(NT, BM), NE), 256, 0, stream>>>(
      h2b, Dd, 0, 0, we1b, FF, (long)Dd * FF, 0, hidb, FF, 0, 0,
      be1, FF, 1, NT, FF, Dd, 1.f, perm, nullptr, offs);
  // moe[token] = gate[token] * (hid[slot] @ We2[e] + be2[e])   (scatter store)
  k_gemm<0, 0, 5, 1><<<dim3(CDIV(Dd, BN), CDIV(NT, BM), NE), 256, 0, stream>>>(
      hidb, FF, 0, 0, we2b, Dd, (long)FF * Dd, 0, moef, Dd, 0, 0,
      be2, Dd, 1, NT, Dd, FF, 1.f, perm, gate, offs);

  // ---- h3 = LN(h2 + moe); out = h3 @ Wout ----
  k_ln_res<<<dim3(NT), dim3(256), 0, stream>>>(h2f, moef, ln3g, ln3b, (float*)nullptr, h3b);
  k_gemm<0, 0, 0, 0><<<dim3(CDIV(OUTD, BN), CDIV(NT, BM), 1), 256, 0, stream>>>(
      h3b, Dd, 0, 0, woutb, OUTD, 0, 0, out, OUTD, 0, 0,
      nullptr, 0, 1, NT, OUTD, Dd, 1.f, nullptr, nullptr, nullptr);
}
